// GaugeTokenFoam_35046933136038
// MI455X (gfx1250) — matrix operand order repaired and checked by reading, not compiled, not run on hardware
//
#include <hip/hip_runtime.h>

// ---------------------------------------------------------------------------
// GaugeTokenFoam for MI455X (gfx1250, wave32)
//   K0: E -> bf16 (workspace)
//   K1: sequential foam scan, one workgroup (32 waves == 32 bubbles), LDS-resident
//       emits U[t][n][64] (bf16) and w[t][n] (f32) to workspace
//   K2: logits[t,v] = sum_n w[t,n] * (E[v] . U[t,n])^2 via v_wmma_f32_16x16x32_bf16
//       8 tokens/block: A fragments kept in registers, U staged in padded LDS
// Workspace use: E_bf16 (6.43MB) + U_bf16 (2MB) + w (64KB) ~= 8.6 MB
// ---------------------------------------------------------------------------

#define NB      32
#define DD      64
#define EP      65          // K1 padded LDS row stride (bank-conflict free pairwise reads)
#define SEQLEN  512
#define NEQ     5
#define VOCAB   50257
#define TB      8           // tokens per K2 block
#define UP      72          // K2 LDS row stride in bf16 (36 dwords -> conflict-free)

typedef __attribute__((ext_vector_type(16))) __bf16 v16bf;
typedef __attribute__((ext_vector_type(8)))  __bf16 v8bf;
typedef __attribute__((ext_vector_type(8)))  float  v8f;

union V16 { v8bf h[2]; v16bf v; };

static __device__ __forceinline__ unsigned short f2bf(float f) {
    unsigned u = __float_as_uint(f);
    unsigned r = u + 0x7FFFu + ((u >> 16) & 1u);   // round-to-nearest-even
    return (unsigned short)(r >> 16);
}

// ---------------------------------------------------------------------------
// K0: fp32 -> bf16 conversion
// ---------------------------------------------------------------------------
__global__ void cvt_bf16_kernel(const float* __restrict__ in,
                                unsigned short* __restrict__ outb, int n) {
    int i = blockIdx.x * blockDim.x + threadIdx.x;
    if (i < n) outb[i] = f2bf(in[i]);
}

// ---------------------------------------------------------------------------
// K1: sequential foam scan. One block, 1024 threads = 32 waves.
// Wave n owns bubble n; lane handles dims (lane, lane+32).
// ---------------------------------------------------------------------------
__global__ __launch_bounds__(1024) void foam_scan_kernel(
    const float* __restrict__ E, const float* __restrict__ bases,
    const float* __restrict__ pTemp, const float* __restrict__ pTarget,
    const float* __restrict__ pStep, const float* __restrict__ pDecayBase,
    const float* __restrict__ pSens, const int* __restrict__ tokens,
    unsigned short* __restrict__ Ubf, float* __restrict__ wW)
{
    __shared__ float sMem[NB * DD];
    __shared__ float sState[NB * DD];
    __shared__ float sMeas[NB * DD];
    __shared__ float sExpr[NB * EP];
    __shared__ float sExprN[NB * EP];   // reused for forces_shared mid-iteration
    __shared__ float sW[NB * NB];       // mask * interaction
    __shared__ float sFm[NB * 33];      // force_mag / ||diff||  (padded)
    __shared__ float sXwm[DD];
    __shared__ float sMm[DD];
    __shared__ float sNormSq[NB];
    __shared__ float sOd[NB];
    __shared__ float sScal[4];

    const int tid  = threadIdx.x;
    const int n    = tid >> 5;          // wave id == bubble id
    const int lane = tid & 31;
    const int d0   = lane, d1 = lane + 32;

    const float temp      = fmaxf(fabsf(pTemp[0]), 0.01f);
    const float target    = pTarget[0];
    const float stepsz    = fminf(fmaxf(fabsf(pStep[0]), 0.001f), 0.5f);
    const float decayBase = pDecayBase[0];
    const float sens      = fabsf(pSens[0]);

    const float* __restrict__ Bn = bases + n * (DD * DD);

    // ---- one-time: interaction = mask * softmax(-(1 - fn@fn^T)/temp) ----
    {
        const int i = n, j = lane;
        const float* Bi = bases + i * (DD * DD);
        const float* Bj = bases + j * (DD * DD);
        float dot = 0.f;
        for (int k = 0; k < DD * DD; ++k) dot += Bi[k] * Bj[k];
        sW[i * NB + j] = dot;
        __syncthreads();
        float ni = sqrtf(sW[i * NB + i]) + 1e-10f;
        float nj = sqrtf(sW[j * NB + j]) + 1e-10f;
        __syncthreads();                       // all diag reads done before overwrite
        float s = -(1.0f - dot / (ni * nj)) / temp;
        float mx = s;
        for (int m = 16; m >= 1; m >>= 1) mx = fmaxf(mx, __shfl_xor(mx, m, 32));
        float ex = __expf(s - mx);
        float sum = ex;
        for (int m = 16; m >= 1; m >>= 1) sum += __shfl_xor(sum, m, 32);
        sW[i * NB + j] = (i == j) ? 0.f : (ex / sum);
    }
    sMem[n * DD + d0] = 0.f;
    sMem[n * DD + d1] = 0.f;
    __syncthreads();

    for (int t = 0; t < SEQLEN; ++t) {
        const int tok = tokens[t];
        // memory mean + raw token embedding
        if (tid < DD) {
            float s = 0.f;
            for (int b = 0; b < NB; ++b) s += sMem[b * DD + tid];
            sMm[tid]  = s * (1.0f / NB);
            sXwm[tid] = E[(size_t)tok * DD + tid];
        }
        __syncthreads();
        if (tid == 0) {
            float dxm = 0.f, mm2 = 0.f, x2 = 0.f;
            for (int d = 0; d < DD; ++d) {
                float m = sMm[d], x = sXwm[d];
                dxm += x * m; mm2 += m * m; x2 += x * x;
            }
            float mem_n = sqrtf(mm2) + 1e-10f;
            float x_n   = sqrtf(x2) + 1e-10f;
            float novelty = (mem_n > 1e-8f) ? (1.0f - dxm / (x_n * mem_n)) : 1.0f;
            float z = decayBase - sens * novelty;
            sScal[0] = 1.0f / (1.0f + __expf(-z));
        }
        __syncthreads();
        const float decay = sScal[0];
        if (tid < DD) sXwm[tid] += decay * sMm[tid];
        __syncthreads();

        // meas[n][e] = sum_d xwm[d] * B[n][d][e]; state = meas
        {
            float m0 = 0.f, m1 = 0.f;
            for (int d = 0; d < DD; ++d) {
                float xv = sXwm[d];
                m0 += xv * Bn[d * DD + d0];
                m1 += xv * Bn[d * DD + d1];
            }
            sMeas[n * DD + d0] = m0; sState[n * DD + d0] = m0;
            sMeas[n * DD + d1] = m1; sState[n * DD + d1] = m1;
        }
        __syncthreads();

        // equilibrium iterations (final pass only recomputes expr/expr_n)
        for (int it = 0; it <= NEQ; ++it) {
            float e0 = 0.f, e1 = 0.f;
            for (int e = 0; e < DD; ++e) {
                float sv = sState[n * DD + e];
                e0 += sv * Bn[d0 * DD + e];
                e1 += sv * Bn[d1 * DD + e];
            }
            sExpr[n * EP + d0] = e0; sExpr[n * EP + d1] = e1;
            float sq = e0 * e0 + e1 * e1;
            for (int m = 16; m >= 1; m >>= 1) sq += __shfl_xor(sq, m, 32);
            float invn = 1.0f / (sqrtf(sq) + 1e-10f);
            sExprN[n * EP + d0] = e0 * invn;
            sExprN[n * EP + d1] = e1 * invn;
            if (it == NEQ) {
                if (lane == 0) sNormSq[n] = sq * invn * invn;  // ||expr_n||^2 for trace
                __syncthreads();
                break;
            }
            __syncthreads();

            // pairwise: cos_sim on expr_n, diff norm on expr
            {
                const int i = n, j = lane;
                float dotp = 0.f, dsq = 0.f;
                for (int d = 0; d < DD; ++d) {
                    dotp += sExprN[i * EP + d] * sExprN[j * EP + d];
                    float dd = sExpr[i * EP + d] - sExpr[j * EP + d];
                    dsq += dd * dd;
                }
                float fm = (dotp - target) * sW[i * NB + j];
                sFm[i * 33 + j] = fm / (sqrtf(dsq) + 1e-10f);
            }
            __syncthreads();

            // forces_shared[n][d] = sum_m fm[n][m]*(expr[n][d]-expr[m][d]) -> reuse sExprN
            {
                float f0 = 0.f, f1 = 0.f;
                float en0 = sExpr[n * EP + d0], en1 = sExpr[n * EP + d1];
                for (int m = 0; m < NB; ++m) {
                    float c = sFm[n * 33 + m];
                    f0 += c * (en0 - sExpr[m * EP + d0]);
                    f1 += c * (en1 - sExpr[m * EP + d1]);
                }
                sExprN[n * EP + d0] = f0;
                sExprN[n * EP + d1] = f1;
            }
            __syncthreads();

            // forces_local[n][e] = sum_d fs[n][d] * B[n][d][e]; state update
            {
                float l0 = 0.f, l1 = 0.f;
                for (int d = 0; d < DD; ++d) {
                    float fv = sExprN[n * EP + d];
                    l0 += fv * Bn[d * DD + d0];
                    l1 += fv * Bn[d * DD + d1];
                }
                sState[n * DD + d0] += stepsz * l0;
                sState[n * DD + d1] += stepsz * l1;
            }
            __syncthreads();
        }

        // energies -> softmax -> weights w[t][n] = od[n] / (tr + eps)
        {
            float q0 = sState[n * DD + d0] - sMeas[n * DD + d0];
            float q1 = sState[n * DD + d1] - sMeas[n * DD + d1];
            float en = q0 * q0 + q1 * q1;
            for (int m = 16; m >= 1; m >>= 1) en += __shfl_xor(en, m, 32);
            if (lane == 0) sOd[n] = en;
        }
        __syncthreads();
        if (n == 0) {
            float s = -sOd[lane] / temp;
            float mx = s;
            for (int m = 16; m >= 1; m >>= 1) mx = fmaxf(mx, __shfl_xor(mx, m, 32));
            float ex = __expf(s - mx);
            float sum = ex;
            for (int m = 16; m >= 1; m >>= 1) sum += __shfl_xor(sum, m, 32);
            float od = ex / sum;
            float trp = od * sNormSq[lane];
            for (int m = 16; m >= 1; m >>= 1) trp += __shfl_xor(trp, m, 32);
            wW[t * NB + lane] = od / (trp + 1e-10f);
        }
        // emit U[t] = expr_n (bf16), row-per-bubble
        {
            unsigned short* Ut = Ubf + (size_t)t * (NB * DD);
            Ut[n * DD + d0] = f2bf(sExprN[n * EP + d0]);
            Ut[n * DD + d1] = f2bf(sExprN[n * EP + d1]);
        }
        // memory = decay*memory + (1-decay)*eq
        {
            float b = 1.0f - decay;
            sMem[n * DD + d0] = decay * sMem[n * DD + d0] + b * sState[n * DD + d0];
            sMem[n * DD + d1] = decay * sMem[n * DD + d1] + b * sState[n * DD + d1];
        }
        __syncthreads();
    }
}

// ---------------------------------------------------------------------------
// K2: logits[t,v] = sum_n w[t,n] * (E[v] . U[t,n])^2
// Block: 8 waves x 128 rows x 8 tokens. A fragments live in registers across
// tokens; U for all 8 tokens staged once in padded LDS (bank-conflict free).
// 32x v_wmma_f32_16x16x32_bf16 per wave, f32 accumulate, fused square+reduce.
// ---------------------------------------------------------------------------
__global__ __launch_bounds__(256) void born_logits_kernel(
    const __bf16* __restrict__ Ebf, const __bf16* __restrict__ Ubf,
    const float* __restrict__ wW, float* __restrict__ out)
{
    __shared__ __bf16 sU[TB * NB * UP];   // 8 tokens x 32 rows x 72 bf16 = 36 KB

    const int wave = threadIdx.x >> 5;
    const int lane = threadIdx.x & 31;
    const int nIdx = lane & 15;       // column-within-tile / A-row-within-tile
    const int hi   = lane >> 4;       // lane half (K-group selector)
    const int tBase = blockIdx.y * TB;
    const int mBase = blockIdx.x * 128 + wave * 16;

    int rowA = mBase + nIdx;
    if (rowA >= VOCAB) rowA = VOCAB - 1;   // clamp loads; stores guarded below

    // ---- cooperative stage of 8 tokens' U into padded LDS (one row/thread) ----
    {
        const int r = threadIdx.x;                           // token*32 + bubble
        const __bf16* __restrict__ src = Ubf + ((size_t)tBase * NB + r) * DD;
        __bf16* dst = &sU[r * UP];
#pragma unroll
        for (int q = 0; q < 4; ++q) {
            *(v8bf*)(dst + q * 16)     = *(const v8bf*)(src + q * 16);
            *(v8bf*)(dst + q * 16 + 8) = *(const v8bf*)(src + q * 16 + 8);
        }
    }

    // ---- A fragments (registers, reused across all 8 tokens) ----
    // A (16x32 bf16): lanes 0-15 M=lane, elems 0-7 -> K=ko..+7, 8-15 -> K=ko+16..+23
    //                 lanes 16-31: K groups shifted by +8
    V16 a0, a1;
    {
        const __bf16* __restrict__ Arow = Ebf + (size_t)rowA * DD;
        const __bf16* pa0 = Arow + 0 + 8 * hi;
        a0.h[0] = *(const v8bf*)(pa0);
        a0.h[1] = *(const v8bf*)(pa0 + 16);
        const __bf16* pa1 = Arow + 32 + 8 * hi;
        a1.h[0] = *(const v8bf*)(pa1);
        a1.h[1] = *(const v8bf*)(pa1 + 16);
    }
    __syncthreads();

    for (int tt = 0; tt < TB; ++tt) {
        const int t = tBase + tt;
        const __bf16* Ut = &sU[tt * NB * UP];

        v8f acc0 = {}; v8f acc1 = {};
#pragma unroll
        for (int kk = 0; kk < 2; ++kk) {
            const int ko = kk * 32;
            // B (32x16 bf16): lane N = nIdx, elems 0-7 -> K = ko + 8*hi + (0..7), 8-15 -> K = ko + 16 + 8*hi + (0..7)
            V16 b0, b1;
            const __bf16* pb0 = Ut + nIdx * UP + ko + 8 * hi;
            b0.h[0] = *(const v8bf*)(pb0);
            b0.h[1] = *(const v8bf*)(pb0 + 16);
            const __bf16* pb1 = Ut + (16 + nIdx) * UP + ko + 8 * hi;
            b1.h[0] = *(const v8bf*)(pb1);
            b1.h[1] = *(const v8bf*)(pb1 + 16);
            const v16bf av = kk ? a1.v : a0.v;
            acc0 = __builtin_amdgcn_wmma_f32_16x16x32_bf16(false, av, false, b0.v,
                                                           (short)0, acc0, false, false);
            acc1 = __builtin_amdgcn_wmma_f32_16x16x32_bf16(false, av, false, b1.v,
                                                           (short)0, acc1, false, false);
        }

        const float w0 = wW[t * NB + nIdx];
        const float w1 = wW[t * NB + 16 + nIdx];
        float part[8];
#pragma unroll
        for (int v = 0; v < 8; ++v) {
            // lane holds D[M = v + 8*hi][N = nIdx (+16 for acc1)]
            float p = w0 * acc0[v] * acc0[v] + w1 * acc1[v] * acc1[v];
            for (int m = 8; m >= 1; m >>= 1) p += __shfl_xor(p, m, 16); // sum 32 cols
            part[v] = p;
        }
        if (nIdx == 0) {
#pragma unroll
            for (int v = 0; v < 8; ++v) {
                int r = mBase + v + 8 * hi;
                if (r < VOCAB) out[(size_t)t * VOCAB + r] = part[v];
            }
        }
    }
}

// ---------------------------------------------------------------------------
extern "C" void kernel_launch(void* const* d_in, const int* in_sizes, int n_in,
                              void* d_out, int out_size, void* d_ws, size_t ws_size,
                              hipStream_t stream) {
    const float* E          = (const float*)d_in[0];
    const float* bases      = (const float*)d_in[1];
    const float* pTemp      = (const float*)d_in[2];
    const float* pTarget    = (const float*)d_in[3];
    const float* pStep      = (const float*)d_in[4];
    const float* pDecayBase = (const float*)d_in[5];
    const float* pSens      = (const float*)d_in[6];
    const int*   tokens     = (const int*)d_in[7];
    float* out = (float*)d_out;

    char* ws = (char*)d_ws;
    const size_t eElems = (size_t)VOCAB * DD;               // 3,216,448
    unsigned short* Ebf = (unsigned short*)ws;               // 6,432,896 B (32B multiple)
    unsigned short* Ubf = (unsigned short*)(ws + eElems * 2);
    float* wW = (float*)(ws + eElems * 2 + (size_t)SEQLEN * NB * DD * 2);

    // K0: E -> bf16
    int nE = (int)eElems;
    cvt_bf16_kernel<<<(nE + 255) / 256, 256, 0, stream>>>(E, Ebf, nE);

    // K1: sequential foam scan (single workgroup carries the recurrence)
    foam_scan_kernel<<<1, 1024, 0, stream>>>(E, bases, pTemp, pTarget, pStep,
                                             pDecayBase, pSens, tokens, Ubf, wW);

    // K2: Born-rule logits via bf16 WMMA, 8 tokens per block
    dim3 grid((VOCAB + 127) / 128, SEQLEN / TB);
    born_logits_kernel<<<grid, 256, 0, stream>>>((const __bf16*)Ebf,
                                                 (const __bf16*)Ubf, wW, out);
}
